// AttentionWithContext_74637941670042
// MI455X (gfx1250) — matrix operand order repaired and checked by reading, not compiled, not run on hardware
//
#include <hip/hip_runtime.h>
#include <math.h>

// CDNA5 (gfx1250) wave32 WMMA types
typedef __attribute__((ext_vector_type(16))) _Float16 v16h;
typedef __attribute__((ext_vector_type(8)))  _Float16 v8h;
typedef __attribute__((ext_vector_type(8)))  float    v8f;

#define B_DIM 64
#define T_DIM 2048
#define F_DIM 256
#define M_ROWS (B_DIM * T_DIM)   // 131072 GEMM rows
#define EPS_K 1e-7f
#define WS_STRIDE 40             // padded halves per W^T row (80 B, 16-B aligned chunks)

static __device__ __forceinline__ v16h cat8(v8h a, v8h b) {
    return __builtin_shufflevector(a, b, 0, 1, 2, 3, 4, 5, 6, 7,
                                         8, 9, 10, 11, 12, 13, 14, 15);
}

// ---------------------------------------------------------------------------
// Phase 1: ait[row] = dot(u, tanh(x[row,:] @ W + b))
// WG = 256 threads = 8 wave32; each wave owns one 16-row M-tile, all N=256.
// Split-f16 GEMM: A*B ~= Ah*Bh + Ah*Bl + Al*Bh  (f32 accumulate, ~fp32 accuracy)
// All LDS fragment reads are contiguous 16-B ds_load_b128.
// ---------------------------------------------------------------------------
__global__ __launch_bounds__(256)
void gemm_tanh_dot_kernel(const float* __restrict__ x,
                          const float* __restrict__ W,
                          const float* __restrict__ bvec,
                          const float* __restrict__ uvec,
                          float* __restrict__ ait)
{
    // x slab: row-major [128][32]  (A-fragment chunks contiguous along K)
    __shared__ alignas(16) _Float16 xs_hi[128 * 32];
    __shared__ alignas(16) _Float16 xs_lo[128 * 32];
    // W slab: TRANSPOSED [n][k], padded stride (B-fragment chunks contiguous)
    __shared__ alignas(16) _Float16 wt_hi[256 * WS_STRIDE];
    __shared__ alignas(16) _Float16 wt_lo[256 * WS_STRIDE];

    const int tid  = threadIdx.x;
    const int wave = tid >> 5;
    const int lane = tid & 31;
    const int half = lane >> 4;       // 0: lanes 0-15, 1: lanes 16-31
    const int l16  = lane & 15;
    const int row0 = blockIdx.x * 128;

    // Per-lane slices of u and bias: column n = nt*16 + l16 (B/C fragment column)
    float u_l[16], b_l[16];
#pragma unroll
    for (int nt = 0; nt < 16; ++nt) {
        u_l[nt] = uvec[nt * 16 + l16];
        b_l[nt] = bvec[nt * 16 + l16];
    }

    v8f acc[16] = {};   // 16 N-tiles of 16x16 f32 accumulators

    for (int k0 = 0; k0 < 256; k0 += 32) {
        __syncthreads();   // previous K-step's fragment reads done

        // Stage x slab (128 rows x 32 k) as f16 hi + residual lo, row-major.
        for (int i = tid; i < 128 * 32; i += 256) {
            const int r  = i >> 5;
            const int kk = i & 31;
            const float v  = x[(size_t)(row0 + r) * F_DIM + (k0 + kk)];
            const _Float16 hi = (_Float16)v;
            xs_hi[i] = hi;
            xs_lo[i] = (_Float16)(v - (float)hi);
        }
        // Stage W slab transposed: thread = column n, loop k (global coalesced
        // across the WG each iteration; W stays L2-resident across the grid).
        {
            const int n = tid;
#pragma unroll
            for (int kk = 0; kk < 32; ++kk) {
                const float v = W[(size_t)(k0 + kk) * F_DIM + n];
                const _Float16 hi = (_Float16)v;
                wt_hi[n * WS_STRIDE + kk] = hi;
                wt_lo[n * WS_STRIDE + kk] = (_Float16)(v - (float)hi);
            }
        }
        __syncthreads();

        // A fragment (16x32 f16): lane holds row m = l16.
        // halves e=0..7 -> K = e + 8*half ; e=8..15 -> K = (e-8) + 16 + 8*half
        // => two contiguous 16-B chunks at half-offsets 8*half and 16 + 8*half.
        const _Float16* arow_hi = &xs_hi[(wave * 16 + l16) * 32];
        const _Float16* arow_lo = &xs_lo[(wave * 16 + l16) * 32];
        const v16h a_hi = cat8(*(const v8h*)(arow_hi + 8 * half),
                               *(const v8h*)(arow_hi + 16 + 8 * half));
        const v16h a_lo = cat8(*(const v8h*)(arow_lo + 8 * half),
                               *(const v8h*)(arow_lo + 16 + 8 * half));

#pragma unroll
        for (int nt = 0; nt < 16; ++nt) {
            // B fragment (32x16 f16): lane holds column n = nt*16 + l16;
            // halves e=0..7 -> K = e + 8*half ; e=8..15 -> K = (e-8) + 16 + 8*half (matches A).
            const _Float16* bcol_hi = &wt_hi[(nt * 16 + l16) * WS_STRIDE + 8 * half];
            const _Float16* bcol_lo = &wt_lo[(nt * 16 + l16) * WS_STRIDE + 8 * half];
            const v16h b_hi = cat8(*(const v8h*)(bcol_hi),
                                   *(const v8h*)(bcol_hi + 16));
            const v16h b_lo = cat8(*(const v8h*)(bcol_lo),
                                   *(const v8h*)(bcol_lo + 16));

            acc[nt] = __builtin_amdgcn_wmma_f32_16x16x32_f16(
                false, a_hi, false, b_hi, (short)0, acc[nt], false, false);
            acc[nt] = __builtin_amdgcn_wmma_f32_16x16x32_f16(
                false, a_hi, false, b_lo, (short)0, acc[nt], false, false);
            acc[nt] = __builtin_amdgcn_wmma_f32_16x16x32_f16(
                false, a_lo, false, b_hi, (short)0, acc[nt], false, false);
        }
    }

    // Epilogue: C layout -> VGPR r holds row r (lanes 0-15) / row r+8 (lanes 16-31)
#pragma unroll
    for (int r = 0; r < 8; ++r) {
        float s = 0.f;
#pragma unroll
        for (int nt = 0; nt < 16; ++nt)
            s += tanhf(acc[nt][r] + b_l[nt]) * u_l[nt];
        // butterfly within each 16-lane half (offsets touch bits 0-3 only)
        s += __shfl_xor(s, 1, 32);
        s += __shfl_xor(s, 2, 32);
        s += __shfl_xor(s, 4, 32);
        s += __shfl_xor(s, 8, 32);
        if (l16 == 0)
            ait[row0 + wave * 16 + r + half * 8] = s;
    }
}

// ---------------------------------------------------------------------------
// Phase 2a: denom[b] = sum_t exp(ait[b,t])   (no max-subtract, per reference)
// ---------------------------------------------------------------------------
__global__ __launch_bounds__(256)
void softmax_denom_kernel(const float* __restrict__ ait, float* __restrict__ denom)
{
    __shared__ float red[8];
    const int b   = blockIdx.x;
    const int tid = threadIdx.x;
    float s = 0.f;
    for (int t = tid; t < T_DIM; t += 256)
        s += expf(ait[(size_t)b * T_DIM + t]);
    s += __shfl_xor(s, 1, 32);
    s += __shfl_xor(s, 2, 32);
    s += __shfl_xor(s, 4, 32);
    s += __shfl_xor(s, 8, 32);
    s += __shfl_xor(s, 16, 32);
    if ((tid & 31) == 0) red[tid >> 5] = s;
    __syncthreads();
    if (tid == 0) {
        float tot = 0.f;
#pragma unroll
        for (int i = 0; i < 8; ++i) tot += red[i];
        denom[b] = tot;
    }
}

// ---------------------------------------------------------------------------
// Phase 2b: partials[b, chunk, f] = sum over one 256-wide T-chunk of
//           x[b,t,f] * exp(ait[b,t]).  x (128 MB) is L2-resident (192 MB L2).
// Deterministic (no float atomics).
// ---------------------------------------------------------------------------
__global__ __launch_bounds__(256)
void weighted_partial_kernel(const float* __restrict__ x,
                             const float* __restrict__ ait,
                             float* __restrict__ partials)
{
    __shared__ float aw[256];
    const int b     = blockIdx.x >> 3;
    const int chunk = blockIdx.x & 7;
    const int t0    = chunk * 256;
    const int f     = threadIdx.x;

    aw[f] = expf(ait[(size_t)b * T_DIM + t0 + f]);
    __syncthreads();

    float acc = 0.f;
    const float* xp = x + ((size_t)b * T_DIM + t0) * F_DIM + f;
    for (int j = 0; j < 256; ++j)
        acc = fmaf(xp[(size_t)j * F_DIM], aw[j], acc);   // coalesced over f

    partials[((size_t)b * 8 + chunk) * F_DIM + f] = acc;
}

// ---------------------------------------------------------------------------
// Phase 2c: out[b,f] = (sum_c partials[b,c,f]) / (denom[b] + EPS)
// ---------------------------------------------------------------------------
__global__ __launch_bounds__(256)
void finalize_kernel(const float* __restrict__ partials,
                     const float* __restrict__ denom,
                     float* __restrict__ out)
{
    const int b = blockIdx.x;
    const int f = threadIdx.x;
    float s = 0.f;
#pragma unroll
    for (int c = 0; c < 8; ++c)
        s += partials[((size_t)b * 8 + c) * F_DIM + f];
    out[(size_t)b * F_DIM + f] = s / (denom[b] + EPS_K);
}

// ---------------------------------------------------------------------------
extern "C" void kernel_launch(void* const* d_in, const int* in_sizes, int n_in,
                              void* d_out, int out_size, void* d_ws, size_t ws_size,
                              hipStream_t stream)
{
    (void)in_sizes; (void)n_in; (void)out_size; (void)ws_size;
    const float* x = (const float*)d_in[0];   // (64, 2048, 256) f32
    const float* W = (const float*)d_in[1];   // (256, 256)      f32
    const float* b = (const float*)d_in[2];   // (256,)          f32
    const float* u = (const float*)d_in[3];   // (256,)          f32
    float* out = (float*)d_out;               // (64, 256)       f32

    float* ws       = (float*)d_ws;
    float* ait      = ws;                       // 131072 floats
    float* denom    = ws + M_ROWS;              // 64 floats
    float* partials = ws + M_ROWS + 64;         // 64*8*256 floats

    gemm_tanh_dot_kernel<<<M_ROWS / 128, 256, 0, stream>>>(x, W, b, u, ait);
    softmax_denom_kernel<<<B_DIM, 256, 0, stream>>>(ait, denom);
    weighted_partial_kernel<<<B_DIM * 8, 256, 0, stream>>>(x, ait, partials);
    finalize_kernel<<<B_DIM, 256, 0, stream>>>(partials, denom, out);
}
